// DBCNN_30837865185713
// MI455X (gfx1250) — matrix operand order repaired and checked by reading, not compiled, not run on hardware
//
#include <hip/hip_runtime.h>
#include <hip/hip_bf16.h>
#include <math.h>

typedef __attribute__((ext_vector_type(16))) _Float16     v16h;
typedef __attribute__((ext_vector_type(8)))  _Float16     v8h;
typedef __attribute__((ext_vector_type(8)))  float        v8f;
typedef __attribute__((ext_vector_type(4)))  float        v4f;
typedef __attribute__((ext_vector_type(4)))  unsigned int v4u;
typedef __attribute__((ext_vector_type(8)))  int          v8i;
typedef __attribute__((ext_vector_type(4)))  int          v4i;

// ---------------------------------------------------------------------------
// Implicit-GEMM conv (KSxKS, SAME padding) with fused bias / BN affine / ReLU.
// Workgroup = 128 threads (4 waves). Block tile: 64 outCh x 64 pixels.
// Per K-chunk (K=32):
//   * A tile (64oc x 32k f32) is DMA'd global->LDS ONCE per block by wave 0
//     via TENSOR_LOAD_TO_LDS (TDM). Descriptor-driven LDS padding gives a
//     144B row stride (16B-aligned v4f reads, conflict-free banks); TDM OOB
//     zero-fill handles both the K tail and Cout tails (e.g. Cout=48).
//   * B patch (32k x 64px) staged cooperatively into LDS as f16 (coalesced).
//   * Each wave runs 4 x v_wmma_f32_16x16x32_f16 from LDS fragments.
// ---------------------------------------------------------------------------
template<int KS>
__global__ __launch_bounds__(128)
void conv_wmma_kernel(const float* __restrict__ X, const float* __restrict__ Wt,
                      const float* __restrict__ bias, const float* __restrict__ gamma,
                      const float* __restrict__ beta, float* __restrict__ Y,
                      int Cin, int Cout, int H, int Wd, int Ho, int Wo,
                      int stride, int pad, int doRelu)
{
    __shared__ alignas(16) float    As[64][36];   // 64 oc x 32 k + 4 DWORD TDM pad
    __shared__ alignas(32) _Float16 Bs[64][40];   // 32 K x 64 px, padded rows

    const int tid     = threadIdx.x;              // 0..127
    const int lane    = tid & 31;
    const int wave    = tid >> 5;                 // 0..3
    const int img     = blockIdx.z;
    const int ocBlk   = blockIdx.y * 64;
    const int ocTile  = ocBlk + wave * 16;
    const int nBase   = blockIdx.x << 6;
    const int spatial = Ho * Wo;
    const int Ktot    = Cin * KS * KS;
    const int Kc      = (Ktot + 31) >> 5;

    // ---- staging assignment: pixel = tid&63, K-half = (tid>>6)*16 ----
    const int sn  = tid & 63;
    const int kh0 = (tid >> 6) << 4;              // 0 or 16
    const int gn  = nBase + sn;
    const bool gvalid = (gn < spatial);
    const int gnc = gvalid ? gn : 0;
    const int goh = gnc / Wo;
    const int gow = gnc % Wo;

    // ---- fragment index maps (contiguous k order, k = 16*(lane/16) + s in A and B) ----
    const int mRow = lane & 15;                   // A: M row
    const int kbA  = (lane < 16) ? 0 : 16;        // A: K base
    const int colN = lane & 15;                   // B/D: N column
    const int kbB  = (lane < 16) ? 0 : 16;        // B: K base (halfs)

    int  nArr[4];
    bool valid[4];
#pragma unroll
    for (int t = 0; t < 4; ++t) {
        nArr[t]  = nBase + t * 16 + colN;
        valid[t] = (nArr[t] < spatial);
    }

    const bool ocOK = (ocTile < Cout);
    const float* Xim = X + (size_t)img * Cin * H * Wd;

    // Uniform pieces of the TDM descriptor (wave 0 only issues it).
    const unsigned int ldsA = (unsigned int)(uintptr_t)&As[0][0]; // LDS byte offset
    const unsigned int tdim1 = (unsigned int)(Cout - ocBlk);      // remaining oc rows

    v8f zero = {};
    v8f acc[4] = {zero, zero, zero, zero};

    for (int kc = 0; kc < Kc; ++kc) {
        const int k0 = kc << 5;

        // ---- wave 0: TDM load of A tile (64 x 32 f32) into LDS ----
        if (wave == 0) {
            unsigned long long ga =
                (unsigned long long)(uintptr_t)(Wt + (size_t)ocBlk * Ktot + k0);
            // D# group 0: count=1 | lds_addr | global_addr | type=2
            v4u g0 = { 1u,
                       ldsA,
                       (unsigned int)(ga & 0xFFFFFFFFull),
                       (unsigned int)((ga >> 32) & 0x01FFFFFFull) | (2u << 30) };
            // D# group 1: data_size=4B, pad_enable, interval=32 DW, amount=4 DW,
            //             tensor_dim0 = Ktot-k0 (K tail zero-fill),
            //             tensor_dim1 = Cout-ocBlk (oc tail zero-fill),
            //             tile_dim0=32, tile_dim1=64, dim0_stride=Ktot
            unsigned int td0 = (unsigned int)(Ktot - k0);
            v8i g1 = { (int)((2u << 16) | (1u << 20) | (4u << 22) | (3u << 25)),
                       (int)((td0 & 0xFFFFu) << 16),
                       (int)((td0 >> 16) | ((tdim1 & 0xFFFFu) << 16)),
                       (int)((tdim1 >> 16) | (32u << 16)),
                       (int)64,
                       (int)(unsigned int)Ktot,
                       0, 0 };
            v4i gz4 = { 0, 0, 0, 0 };
            v8i gz8 = { 0, 0, 0, 0, 0, 0, 0, 0 };
            __builtin_amdgcn_tensor_load_to_lds(g0, g1, gz4, gz4, gz8, 0);
            __builtin_amdgcn_s_wait_tensorcnt(0);
        }

        // ---- stage B patch (32K x 64px) into LDS as f16 ----
#pragma unroll
        for (int j = 0; j < 16; ++j) {
            int   k  = k0 + kh0 + j;
            float xv = 0.0f;
            if (k < Ktot && gvalid) {
                int ic, kh, kw;
                if (KS == 1) { ic = k; kh = 0; kw = 0; }
                else         { ic = k / 9; int r = k - ic * 9; kh = r / 3; kw = r - kh * 3; }
                int ih = goh * stride - pad + kh;
                int iw = gow * stride - pad + kw;
                if ((unsigned)ih < (unsigned)H && (unsigned)iw < (unsigned)Wd)
                    xv = Xim[(size_t)ic * H * Wd + (size_t)ih * Wd + iw];
            }
            Bs[sn][kh0 + j] = (_Float16)xv;
        }

        __syncthreads();

        // ---- A fragment from LDS (one 16-float run: k=kbA..kbA+15) ----
        const float* arow = &As[wave * 16 + mRow][0];
        v4f a0 = *(const v4f*)(arow + kbA);
        v4f a1 = *(const v4f*)(arow + kbA + 4);
        v4f a2 = *(const v4f*)(arow + kbA + 8);
        v4f a3 = *(const v4f*)(arow + kbA + 12);
        v16h a;
#pragma unroll
        for (int e = 0; e < 4; ++e) {
            a[e]      = (_Float16)a0[e];
            a[e + 4]  = (_Float16)a1[e];
            a[e + 8]  = (_Float16)a2[e];
            a[e + 12] = (_Float16)a3[e];
        }

        // ---- 4 WMMAs, B fragments from LDS ----
#pragma unroll
        for (int t = 0; t < 4; ++t) {
            const _Float16* rowp = &Bs[t * 16 + colN][kbB];
            v8h lo = *(const v8h*)(rowp);
            v8h hi = *(const v8h*)(rowp + 8);
            v16h b = __builtin_shufflevector(lo, hi,
                     0, 1, 2, 3, 4, 5, 6, 7, 8, 9, 10, 11, 12, 13, 14, 15);
            // EXEC is all-ones here (no divergent branch encloses this call).
            acc[t] = __builtin_amdgcn_wmma_f32_16x16x32_f16(
                false, a, false, b, (short)0, acc[t], false, false);
        }

        __syncthreads();
    }

    // ---- epilogue: C/D layout — VGPR r: lanes0-15 M=r, lanes16-31 M=r+8 ----
    if (!ocOK) return;
    const int hiOff = (lane < 16) ? 0 : 8;
    const size_t outBaseImg = (size_t)img * Cout * spatial;
#pragma unroll
    for (int t = 0; t < 4; ++t) {
        if (!valid[t]) continue;
#pragma unroll
        for (int r = 0; r < 8; ++r) {
            int   oc = ocTile + r + hiOff;
            float v  = acc[t][r];
            if (bias)   v += bias[oc];
            if (gamma)  v  = v * gamma[oc] + beta[oc];
            if (doRelu) v  = fmaxf(v, 0.0f);
            Y[outBaseImg + (size_t)oc * spatial + nArr[t]] = v;
        }
    }
}

// ---------------------------------------------------------------------------
// 2x2 max pool, stride 2
// ---------------------------------------------------------------------------
__global__ void maxpool2_kernel(const float* __restrict__ X, float* __restrict__ Y,
                                int C, int H, int Wd, int total)
{
    int i = blockIdx.x * blockDim.x + threadIdx.x;
    if (i >= total) return;
    int Ho = H >> 1, Wo = Wd >> 1;
    int x = i % Wo; int tmp = i / Wo;
    int y = tmp % Ho; tmp /= Ho;
    int c = tmp % C; int n = tmp / C;
    const float* p = X + (((size_t)n * C + c) * H + 2 * y) * Wd + 2 * x;
    Y[i] = fmaxf(fmaxf(p[0], p[1]), fmaxf(p[Wd], p[Wd + 1]));
}

// ---------------------------------------------------------------------------
// Global avg + max pool over HW per (n,c)
// ---------------------------------------------------------------------------
__global__ void gpool_kernel(const float* __restrict__ X, float* __restrict__ avg,
                             float* __restrict__ mx, int HW, int total)
{
    int i = blockIdx.x * blockDim.x + threadIdx.x;
    if (i >= total) return;
    const float* p = X + (size_t)i * HW;
    float s = 0.0f, m = -INFINITY;
    for (int j = 0; j < HW; ++j) { float v = p[j]; s += v; m = fmaxf(m, v); }
    avg[i] = s / (float)HW;
    mx[i]  = m;
}

// ---------------------------------------------------------------------------
// Channel attention: sigmoid(W2·relu(W1·avg) + W2·relu(W1·mx))  -> [N,512]
// ---------------------------------------------------------------------------
__global__ __launch_bounds__(512)
void ca_kernel(const float* __restrict__ W1, const float* __restrict__ W2,
               const float* __restrict__ avg, const float* __restrict__ mx,
               float* __restrict__ out)
{
    __shared__ float hS[128];
    int n = blockIdx.x;
    int t = threadIdx.x;
    if (t < 128) {
        float sa = 0.0f, sm = 0.0f;
        const float* w  = W1 + (size_t)t * 512;
        const float* pa = avg + (size_t)n * 512;
        const float* pm = mx  + (size_t)n * 512;
        for (int c = 0; c < 512; ++c) { sa += w[c] * pa[c]; sm += w[c] * pm[c]; }
        hS[t] = fmaxf(sa, 0.0f) + fmaxf(sm, 0.0f);
    }
    __syncthreads();
    float s = 0.0f;
    const float* w2 = W2 + (size_t)t * 128;
    for (int j = 0; j < 128; ++j) s += w2[j] * hS[j];
    out[(size_t)n * 512 + t] = 1.0f / (1.0f + __expf(-s));
}

// ---------------------------------------------------------------------------
// X1 = vgg * caw (broadcast over HW), then per-position channel L2 normalize
// ---------------------------------------------------------------------------
__global__ void scale_cnorm_kernel(const float* __restrict__ X, const float* __restrict__ w,
                                   float* __restrict__ Y, int C, int HW, int total)
{
    int i = blockIdx.x * blockDim.x + threadIdx.x;
    if (i >= total) return;
    int p = i % HW;
    int n = i / HW;
    const float* xb = X + (size_t)n * C * HW + p;
    const float* wb = w + (size_t)n * C;
    float ss = 0.0f;
    for (int c = 0; c < C; ++c) { float v = xb[(size_t)c * HW] * wb[c]; ss += v * v; }
    float inv = 1.0f / fmaxf(sqrtf(ss), 1e-12f);
    float* yb = Y + (size_t)n * C * HW + p;
    for (int c = 0; c < C; ++c) { float v = xb[(size_t)c * HW] * wb[c]; yb[(size_t)c * HW] = v * inv; }
}

// ---------------------------------------------------------------------------
// Deterministic global L2 norm: 32 partials/sample, then fixed-order tree
// ---------------------------------------------------------------------------
__global__ __launch_bounds__(256)
void norm_partial_kernel(const float* __restrict__ X, float* __restrict__ part, int F)
{
    __shared__ float red[256];
    int n     = blockIdx.x >> 5;
    int chunk = blockIdx.x & 31;
    int tid   = threadIdx.x;
    float s = 0.0f;
    for (int i = chunk * 256 + tid; i < F; i += 32 * 256) {
        float v = X[(size_t)n * F + i]; s += v * v;
    }
    red[tid] = s; __syncthreads();
    for (int off = 128; off > 0; off >>= 1) {
        if (tid < off) red[tid] += red[tid + off];
        __syncthreads();
    }
    if (tid == 0) part[blockIdx.x] = red[0];
}

__global__ void norm_final_kernel(const float* __restrict__ part, float* __restrict__ norms)
{
    __shared__ float red[32];
    int n = blockIdx.x, tid = threadIdx.x;
    red[tid] = part[n * 32 + tid]; __syncthreads();
    for (int off = 16; off > 0; off >>= 1) {
        if (tid < off) red[tid] += red[tid + off];
        __syncthreads();
    }
    if (tid == 0) norms[n] = sqrtf(red[0]);
}

// ---------------------------------------------------------------------------
// Soft histogram: v normalized by global norm, 5 bins, feature-major layout
// ---------------------------------------------------------------------------
__global__ void hist_kernel(const float* __restrict__ X, const float* __restrict__ norms,
                            float* __restrict__ hist, int F, int total)
{
    int i = blockIdx.x * blockDim.x + threadIdx.x;
    if (i >= total) return;
    int n = i / F;
    int f = i - n * F;
    float inv = 1.0f / fmaxf(norms[n], 1e-12f);
    float v   = X[i] * inv;
    const float delta = 0.2f, sigma = 15.0f;
    float* hb = hist + ((size_t)n * F + f) * 5;
#pragma unroll
    for (int b = 0; b < 5; ++b) {
        float c = ((float)b + 0.5f) * delta;
        float d = v - c;
        float h = 1.0f / (1.0f + __expf(-sigma * (d + 0.5f * delta)))
                - 1.0f / (1.0f + __expf(-sigma * (d - 0.5f * delta)));
        hb[b] = h;
    }
}

// ---------------------------------------------------------------------------
// FC: logits[n,o] = hist[n,:] . Wfc[o,:] + b[o]  (deterministic tree reduce)
// ---------------------------------------------------------------------------
__global__ __launch_bounds__(256)
void fc_kernel(const float* __restrict__ hist, const float* __restrict__ Wfc,
               const float* __restrict__ bfc, float* __restrict__ logits, int F5)
{
    __shared__ float red[256];
    int n = blockIdx.x / 5, o = blockIdx.x % 5;
    int tid = threadIdx.x;
    const float* h = hist + (size_t)n * F5;
    const float* w = Wfc  + (size_t)o * F5;
    float s = 0.0f;
    for (int i = tid; i < F5; i += 256) s += h[i] * w[i];
    red[tid] = s; __syncthreads();
    for (int off = 128; off > 0; off >>= 1) {
        if (tid < off) red[tid] += red[tid + off];
        __syncthreads();
    }
    if (tid == 0) logits[n * 5 + o] = red[0] + bfc[o];
}

__global__ void softmax_kernel(const float* __restrict__ logits, float* __restrict__ out, int N)
{
    int n = threadIdx.x;
    if (n >= N) return;
    float m = -INFINITY;
    for (int o = 0; o < 5; ++o) m = fmaxf(m, logits[n * 5 + o]);
    float e[5], s = 0.0f;
    for (int o = 0; o < 5; ++o) { e[o] = __expf(logits[n * 5 + o] - m); s += e[o]; }
    for (int o = 0; o < 5; ++o) out[n * 5 + o] = e[o] / s;
}

// ---------------------------------------------------------------------------
// Host-side conv launcher (SAME padding: s=1 -> pad 1; s=2, even H -> pad 0)
// ---------------------------------------------------------------------------
static void launch_conv(int KS, const float* X, const float* W, const float* b,
                        const float* g, const float* be, float* Y,
                        int Cin, int Cout, int H, int Wd, int stride, int relu,
                        int N, hipStream_t s)
{
    int Ho  = (stride == 1) ? H  : H  / 2;
    int Wo  = (stride == 1) ? Wd : Wd / 2;
    int pad = (KS == 3) ? ((stride == 1) ? 1 : 0) : 0;
    dim3 grid((Ho * Wo + 63) / 64, (Cout + 63) / 64, N);
    if (KS == 3)
        conv_wmma_kernel<3><<<grid, 128, 0, s>>>(X, W, b, g, be, Y, Cin, Cout, H, Wd, Ho, Wo, stride, pad, relu);
    else
        conv_wmma_kernel<1><<<grid, 128, 0, s>>>(X, W, b, g, be, Y, Cin, Cout, H, Wd, Ho, Wo, stride, pad, relu);
}

// ---------------------------------------------------------------------------
// Input order (setup_inputs insertion order, depth-first):
//  [0] X  (N,3,224,224)
//  [1..26]  vgg: 13 x (w,b)
//  [27..62] scnn: 9 x (w,b,g,be)
//  [63..66] proj: w,b,g,be
//  [67..68] ca:   w1,w2
//  [69..70] fc:   Wfc,bfc
// ---------------------------------------------------------------------------
extern "C" void kernel_launch(void* const* d_in, const int* in_sizes, int n_in,
                              void* d_out, int out_size, void* d_ws, size_t ws_size,
                              hipStream_t stream)
{
    (void)n_in; (void)out_size; (void)ws_size;
    const float* X = (const float*)d_in[0];
    const int N = in_sizes[0] / (3 * 224 * 224);

    // ---- workspace carve (floats) ----
    float* ws = (float*)d_ws;
    const size_t CAPB = (size_t)N * 64 * 224 * 224;   // largest activation tensor
    float* bufA = ws;
    float* bufB = ws + CAPB;
    float* sm   = ws + 2 * CAPB;
    float* X2   = sm;                                  // N*512*14*14
    float* X2P  = X2  + (size_t)N * 512 * 14 * 14;     // N*512*7*7
    float* AVG  = X2P + (size_t)N * 512 * 7 * 7;       // N*512
    float* MX   = AVG + (size_t)N * 512;               // N*512
    float* CAW  = MX  + (size_t)N * 512;               // N*512
    float* PART = CAW + (size_t)N * 512;               // N*32
    float* NRM  = PART + (size_t)N * 32;               // N
    float* LGT  = NRM + N;                             // N*5
    float* HIST = LGT + (size_t)N * 5;                 // N*512*14*14*5

    // ================= SCNN branch =================
    static const int scnnCin[9]  = {3, 48, 48, 64, 64, 64, 64, 128, 128};
    static const int scnnCout[9] = {48, 48, 64, 64, 64, 64, 128, 128, 128};
    static const int scnnStr[9]  = {1, 2, 1, 2, 1, 2, 1, 1, 2};

    const float* cur = X;
    float* dst = bufA;
    int H = 224;
    for (int j = 0; j < 9; ++j) {
        launch_conv(3, cur,
                    (const float*)d_in[27 + 4 * j], (const float*)d_in[28 + 4 * j],
                    (const float*)d_in[29 + 4 * j], (const float*)d_in[30 + 4 * j],
                    dst, scnnCin[j], scnnCout[j], H, H, scnnStr[j], 1, N, stream);
        if (scnnStr[j] == 2) H /= 2;
        cur = dst;
        dst = (dst == bufA) ? bufB : bufA;
    }
    // proj 1x1 128->512 @14x14, BN+ReLU
    launch_conv(1, cur, (const float*)d_in[63], (const float*)d_in[64],
                (const float*)d_in[65], (const float*)d_in[66],
                X2, 128, 512, 14, 14, 1, 1, N, stream);
    {
        int tot = N * 512 * 7 * 7;
        maxpool2_kernel<<<(tot + 255) / 256, 256, 0, stream>>>(X2, X2P, 512, 14, 14, tot);
        int nc = N * 512;
        gpool_kernel<<<(nc + 255) / 256, 256, 0, stream>>>(X2P, AVG, MX, 49, nc);
        ca_kernel<<<N, 512, 0, stream>>>((const float*)d_in[67], (const float*)d_in[68],
                                         AVG, MX, CAW);
    }

    // ================= VGG16 branch =================
    static const int vggCin[13]  = {3, 64, 64, 128, 128, 256, 256, 256, 512, 512, 512, 512, 512};
    static const int vggCout[13] = {64, 64, 128, 128, 256, 256, 256, 512, 512, 512, 512, 512, 512};

    cur = X;
    dst = bufA;
    H = 224;
    for (int i = 0; i < 13; ++i) {
        launch_conv(3, cur, (const float*)d_in[1 + 2 * i], (const float*)d_in[2 + 2 * i],
                    nullptr, nullptr, dst, vggCin[i], vggCout[i], H, H, 1, 1, N, stream);
        cur = dst;
        dst = (cur == bufA) ? bufB : bufA;
        if (i == 1 || i == 3 || i == 6 || i == 9) {
            int Ho  = H / 2;
            int tot = N * vggCout[i] * Ho * Ho;
            maxpool2_kernel<<<(tot + 255) / 256, 256, 0, stream>>>(cur, dst, vggCout[i], H, H, tot);
            H   = Ho;
            cur = dst;
            dst = (cur == bufA) ? bufB : bufA;
        }
    }

    // ========== attention scaling + normalizations ==========
    const int HW = 14 * 14;
    const int F  = 512 * HW;
    {
        int totp = N * HW;
        scale_cnorm_kernel<<<(totp + 255) / 256, 256, 0, stream>>>(cur, CAW, dst, 512, HW, totp);
    }
    float* X1n = dst;
    norm_partial_kernel<<<N * 32, 256, 0, stream>>>(X1n, PART, F);
    norm_final_kernel<<<N, 32, 0, stream>>>(PART, NRM);

    // ========== soft histogram + FC + softmax ==========
    {
        int toth = N * F;
        hist_kernel<<<(toth + 255) / 256, 256, 0, stream>>>(X1n, NRM, HIST, F, toth);
    }
    fc_kernel<<<N * 5, 256, 0, stream>>>(HIST, (const float*)d_in[69], (const float*)d_in[70],
                                         LGT, F * 5);
    softmax_kernel<<<1, 32, 0, stream>>>(LGT, (float*)d_out, N);
}
